// CHMM_69999376990781
// MI455X (gfx1250) — compile-verified, matrix operand order checked by reading, not run on hardware
//
#include <hip/hip_runtime.h>
#include <hip/hip_bf16.h>
#include <cstdint>

// Problem constants
#define BB 32
#define TT 512
#define DD 768
#define SS 10
#define NMAT 11           // 1 trans + 10 emiss matrices, each [768 x 256]
#define NCT (NMAT * 16)   // 176 column tiles of 16
#define NROWT ((BB * TT) / 16)  // 1024 row tiles

typedef __attribute__((ext_vector_type(16))) _Float16 v16h;
typedef __attribute__((ext_vector_type(8)))  _Float16 v8h;
typedef __attribute__((ext_vector_type(8)))  float    v8f;

// ---- workspace layout (bytes) ----
#define OFF_WT      ((size_t)0)          // f16 packed weights: 11*256*768*2 = 4,325,376
#define OFF_BIAS    ((size_t)4325376)    // 11*256 f32
#define OFF_TB      ((size_t)4336640)    // trans base softmax 16*16 f32
#define OFF_EB      ((size_t)4337664)    // emiss base softmax 10*16*16 f32
#define OFF_LSP     ((size_t)4347904)    // log state priors 16 f32
#define OFF_LOBS    ((size_t)4348032)    // log(obs substituted) B*T*10*16 f32
#define OFF_LEV     ((size_t)14833792)   // log evidence B*T*16 f32
#define OFF_ALPHA   ((size_t)15882368)   // B*T*16 f32
#define OFF_BETA    ((size_t)16930944)   // B*T*16 f32
#define OFF_CONTRIB ((size_t)17979520)   // B*T f32 per-(b,t) ll contributions

#if defined(__HIP_DEVICE_COMPILE__) && __has_builtin(__builtin_amdgcn_tensor_load_to_lds)
#define USE_TDM 1
#else
#define USE_TDM 0
#endif

#if USE_TDM
typedef unsigned int uint32x4 __attribute__((ext_vector_type(4)));
typedef int int32x8 __attribute__((ext_vector_type(8)));
typedef int int32x4 __attribute__((ext_vector_type(4)));
#endif

// ---- DPP16 butterfly reductions: pure VALU, no LDS round-trips ----
// quad_perm(1,0,3,2)=0xB1, quad_perm(2,3,0,1)=0x4E, row_half_mirror=0x141,
// row_mirror=0x140 combine all 16 lanes of a row; rows stay independent.
template <int CTRL>
__device__ __forceinline__ float fdpp(float x) {
  int xi = __float_as_int(x);
  int r = __builtin_amdgcn_update_dpp(xi, xi, CTRL, 0xf, 0xf, true);
  return __int_as_float(r);
}
__device__ __forceinline__ float rmax16(float v) {
  v = fmaxf(v, fdpp<0xB1>(v));
  v = fmaxf(v, fdpp<0x4E>(v));
  v = fmaxf(v, fdpp<0x141>(v));
  v = fmaxf(v, fdpp<0x140>(v));
  return v;
}
__device__ __forceinline__ float rsum16(float v) {
  v += fdpp<0xB1>(v);
  v += fdpp<0x4E>(v);
  v += fdpp<0x141>(v);
  v += fdpp<0x140>(v);
  return v;
}
__device__ __forceinline__ float rmax32(float v) {
  v = rmax16(v);
  return fmaxf(v, __shfl_xor(v, 16, 32));  // single cross-row combine
}
__device__ __forceinline__ float rsum32(float v) {
  v = rsum16(v);
  return v + __shfl_xor(v, 16, 32);
}
// SGPR broadcast of lane i's value (i must be uniform/constant)
__device__ __forceinline__ float bcastlane(float v, int lane) {
  return __int_as_float(__builtin_amdgcn_readlane(__float_as_int(v), lane));
}

// ---- K1: pack W_trans/W_emiss -> f16, column-major [m][n][k] (k contiguous) ----
__global__ __launch_bounds__(256)
void chmm_pack_w(const float* __restrict__ Wt, const float* __restrict__ We,
                 _Float16* __restrict__ wt) {
  size_t idx = (size_t)blockIdx.x * 256 + threadIdx.x;
  if (idx >= (size_t)NMAT * 256 * DD) return;
  int m = (int)(idx / (256 * DD));
  int r = (int)(idx % (256 * DD));
  int n = r / DD, k = r % DD;
  float v = (m == 0) ? Wt[(size_t)k * 256 + n]
                     : We[((size_t)(m - 1) * DD + k) * 256 + n];
  wt[idx] = (_Float16)v;
}

// ---- K2: biases, base softmax(unnorm_trans/emiss), log_softmax(priors) ----
__global__ __launch_bounds__(256)
void chmm_prep_small(const float* __restrict__ b_trans, const float* __restrict__ b_emiss,
                     const float* __restrict__ u_trans, const float* __restrict__ u_emiss,
                     const float* __restrict__ priors,
                     float* __restrict__ bias, float* __restrict__ tb,
                     float* __restrict__ eb, float* __restrict__ lsp) {
  const int tid = threadIdx.x;
  for (int idx = tid; idx < NMAT * 256; idx += 256) {
    int m = idx / 256, n = idx % 256;
    bias[idx] = (m == 0) ? b_trans[n] : b_emiss[(m - 1) * 256 + n];
  }
  if (tid < 176) {  // 16 trans rows + 160 emiss rows, softmax over last dim
    const float* src; float* dst;
    if (tid < 16) { src = u_trans + tid * 16; dst = tb + tid * 16; }
    else          { src = u_emiss + (tid - 16) * 16; dst = eb + (tid - 16) * 16; }
    float mx = src[0];
    for (int i = 1; i < 16; ++i) mx = fmaxf(mx, src[i]);
    float s = 0.f;
    for (int i = 0; i < 16; ++i) s += __expf(src[i] - mx);
    for (int i = 0; i < 16; ++i) dst[i] = __expf(src[i] - mx) / s;
  }
  if (tid == 192) {  // log_softmax(state_priors)
    float mx = priors[0];
    for (int i = 1; i < 16; ++i) mx = fmaxf(mx, priors[i]);
    float s = 0.f;
    for (int i = 0; i < 16; ++i) s += __expf(priors[i] - mx);
    float l = mx + __logf(s);
    for (int i = 0; i < 16; ++i) lsp[i] = priors[i] - l;
  }
}

// ---- K3: observation substitution + log ----
__global__ __launch_bounds__(256)
void chmm_obs_prep(const float* __restrict__ obs, float* __restrict__ lobs) {
  int bt = blockIdx.x * 256 + threadIdx.x;
  if (bt >= BB * TT) return;
  const float* op = obs + (size_t)bt * SS * 16;
  float* lp = lobs + (size_t)bt * SS * 16;
  int lbs[SS]; int sum = 0;
#pragma unroll
  for (int s = 0; s < SS; ++s) {
    const float* row = op + s * 16;
    int bi = 0; float bv = row[0];
    for (int o = 1; o < 16; ++o) if (row[o] > bv) { bv = row[o]; bi = o; }
    lbs[s] = bi; sum += bi;
  }
  bool entity = (sum > 0);
#pragma unroll
  for (int s = 0; s < SS; ++s) {
    bool sub = entity && (lbs[s] == 0);
    const float* row = op + s * 16;
    for (int o = 0; o < 16; ++o) {
      float v = sub ? (o == 0 ? 0.01f : (0.99f / 16.0f)) : row[o];
      lp[s * 16 + o] = __logf(v);
    }
  }
}

// ---- K4: fused WMMA GEMM + row softmax + log-mix + evidence logsumexp ----
// TDM stages the fp32 emb tile into LDS (async DMA, TENSORcnt), then it is
// converted to f16. Wave w handles col tiles ct = w, w+8, ... => emiss h in
// {w, w+8}: evidence accumulates in registers, no atomics.
__global__ __launch_bounds__(256)
void chmm_gemm_fused(const float* __restrict__ emb, const _Float16* __restrict__ wt,
                     const float* __restrict__ bias, const float* __restrict__ tb,
                     const float* __restrict__ eb, const float* __restrict__ lobs,
                     float* __restrict__ out_trans, float* __restrict__ out_emiss,
                     float* __restrict__ lev) {
  __shared__ _Float16 sA[16 * 776];  // f16 emb tile, padded vs bank conflicts
  __shared__ float sObs[16 * 160];   // log obs for these 16 rows: [row][s][o]
#if USE_TDM
  __shared__ float sStage[16 * DD];  // fp32 staging for tensor_load_to_lds
#endif
  const int tid = threadIdx.x;
  const int r0 = blockIdx.x * 16;

#if USE_TDM
  if (tid < 32) {  // wave 0 issues one TDM op for the whole 16x768 f32 tile
    unsigned long long ga = (unsigned long long)(uintptr_t)(emb + (size_t)r0 * DD);
    unsigned int laddr = (unsigned int)(uintptr_t)sStage;  // LDS byte offset (addr[31:0])
    // D# group0: count=1 | lds_addr | global_addr[56:0] | type=2
    uint32x4 g0 = {1u, laddr, (unsigned int)ga,
                   (unsigned int)(ga >> 32) | (2u << 30)};
    // D# group1: data_size=4B; tensor_dim0=768, tensor_dim1=16384;
    // tile 768x16; tensor_dim0_stride=768 (fields per ISA Table 8.4)
    int32x8 g1 = {0x20000, 0x03000000, 0x40000000, 0x03000000, 16, 768, 0, 0};
    int32x4 gz = {0, 0, 0, 0};
#if __clang_major__ >= 23
    int32x8 gz8 = {0, 0, 0, 0, 0, 0, 0, 0};
    __builtin_amdgcn_tensor_load_to_lds(g0, g1, gz, gz, gz8, 0);
#else
    __builtin_amdgcn_tensor_load_to_lds(g0, g1, gz, gz, 0);
#endif
    __builtin_amdgcn_s_wait_tensorcnt(0);
  }
  __syncthreads();
  for (int idx = tid; idx < 16 * DD; idx += 256) {
    int row = idx / DD, k = idx % DD;
    sA[row * 776 + k] = (_Float16)sStage[row * DD + k];
  }
#else
  for (int idx = tid; idx < 16 * DD; idx += 256) {
    int row = idx / DD, k = idx % DD;
    sA[row * 776 + k] = (_Float16)emb[(size_t)(r0 + row) * DD + k];
  }
#endif
  for (int idx = tid; idx < 16 * 160; idx += 256)
    sObs[idx] = lobs[(size_t)r0 * 160 + idx];
  __syncthreads();

  const int wv = tid >> 5, lane = tid & 31, half = lane >> 4, lq = lane & 15;
  const _Float16* ap0 = sA + lq * 776 + half * 8;  // A frag: lane row=lq, K per ISA layout

  float evacc0[8] = {0, 0, 0, 0, 0, 0, 0, 0};  // evidence for h = wv
  float evacc1[8] = {0, 0, 0, 0, 0, 0, 0, 0};  // evidence for h = wv + 8

  for (int ct = wv; ct < NCT; ct += 8) {
    const int m = ct >> 4, cm = ct & 15;
    const float basev = (m == 0) ? tb[cm * 16 + lq]
                                 : eb[((m - 1) * 16 + cm) * 16 + lq];
    const float bv = bias[m * 256 + cm * 16 + lq];
    const _Float16* wp = wt + (size_t)(m * 256 + cm * 16 + lq) * DD + half * 8;

    v8f c = {0.f, 0.f, 0.f, 0.f, 0.f, 0.f, 0.f, 0.f};
#pragma unroll 4
    for (int k0 = 0; k0 < DD; k0 += 32) {
      __builtin_prefetch(wp + k0 + 128, 0, 1);  // -> global_prefetch_b8
      v8h alo = *(const v8h*)(ap0 + k0);
      v8h ahi = *(const v8h*)(ap0 + k0 + 16);
      v8h blo = *(const v8h*)(wp + k0);
      v8h bhi = *(const v8h*)(wp + k0 + 16);
      v16h av = __builtin_shufflevector(alo, ahi, 0, 1, 2, 3, 4, 5, 6, 7,
                                        8, 9, 10, 11, 12, 13, 14, 15);
      v16h bvv = __builtin_shufflevector(blo, bhi, 0, 1, 2, 3, 4, 5, 6, 7,
                                         8, 9, 10, 11, 12, 13, 14, 15);
      c = __builtin_amdgcn_wmma_f32_16x16x32_f16(false, av, false, bvv,
                                                 (short)0, c, false, false);
    }

#pragma unroll
    for (int j = 0; j < 8; ++j) {
      // C/D layout: lane holds col N=lq, row M = j + 8*half
      float v = c[j] + bv;
      float mx = rmax16(v);            // softmax over the 16 tile columns (row-wise)
      float e = __expf(v - mx);
      float s = rsum16(e);
      float nn = e / s;
      float ov = __logf(0.5f * basev + 0.5f * nn);  // log-mix with base matrix
      int row_l = j + 8 * half;
      size_t rg = (size_t)(r0 + row_l);
      if (m == 0) {
        out_trans[rg * 256 + cm * 16 + lq] = ov;
      } else {
        out_emiss[rg * 2560 + (size_t)(m - 1) * 256 + cm * 16 + lq] = ov;
        // evidence: logsumexp over O of (log_emiss + log obs), summed over sources
        float vv = ov + sObs[row_l * 160 + (m - 1) * 16 + lq];
        float m2 = rmax16(vv);
        float se = rsum16(__expf(vv - m2));
        float lse = m2 + __logf(se);
        if (cm < 8) evacc0[j] += lse; else evacc1[j] += lse;
      }
    }
  }
  if (lq == 0) {  // lanes 0 (rows 0-7) and 16 (rows 8-15)
#pragma unroll
    for (int j = 0; j < 8; ++j) {
      int row_l = j + 8 * half;
      lev[(size_t)(r0 + row_l) * 16 + wv] = evacc0[j];
      lev[(size_t)(r0 + row_l) * 16 + wv + 8] = evacc1[j];
    }
  }
}

// ---- K5: forward (blocks 0..31) and backward (blocks 32..63) recursions ----
__global__ __launch_bounds__(32)
void chmm_fb(const float* __restrict__ lt_all, const float* __restrict__ lev,
             const float* __restrict__ lsp, float* __restrict__ alpha,
             float* __restrict__ beta) {
  const int lane = threadIdx.x & 31;
  const int j = lane & 15;
  const int blk = blockIdx.x;
  if (blk < BB) {  // forward
    const int b = blk;
    const float* ev = lev + (size_t)b * TT * 16;
    const float* lt = lt_all + (size_t)b * TT * 256;
    float* al = alpha + (size_t)b * TT * 16;
    float a = lsp[j] + ev[j];
    { float mx = rmax16(a); float s = rsum16(__expf(a - mx)); a -= mx + __logf(s); }
    if (lane < 16) al[j] = a;
    for (int t = 1; t < TT; ++t) {
      const float* ltt = lt + (size_t)t * 256;
      float r[16];
#pragma unroll
      for (int i = 0; i < 16; ++i) r[i] = bcastlane(a, i) + ltt[i * 16 + j];
      float mx = r[0];
#pragma unroll
      for (int i = 1; i < 16; ++i) mx = fmaxf(mx, r[i]);
      float se = 0.f;
#pragma unroll
      for (int i = 0; i < 16; ++i) se += __expf(r[i] - mx);
      a = ev[t * 16 + j] + mx + __logf(se);
      float m2 = rmax16(a); float s2 = rsum16(__expf(a - m2)); a -= m2 + __logf(s2);
      if (lane < 16) al[(size_t)t * 16 + j] = a;
    }
  } else {  // backward (no per-step normalization, faithful to reference)
    const int b = blk - BB;
    const float* ev = lev + (size_t)b * TT * 16;
    const float* lt = lt_all + (size_t)b * TT * 256;
    float* be = beta + (size_t)b * TT * 16;
    float bn = 0.f;
    if (lane < 16) be[(size_t)(TT - 1) * 16 + j] = 0.f;
    for (int t = TT - 2; t >= 0; --t) {
      float v = ev[t * 16 + j] + bn;       // (ev + beta_next)[j], lane j
      const float* ltt = lt + (size_t)t * 256;
      float r[16];
#pragma unroll
      for (int jj = 0; jj < 16; ++jj) r[jj] = ltt[j * 16 + jj] + bcastlane(v, jj);
      float mx = r[0];
#pragma unroll
      for (int jj = 1; jj < 16; ++jj) mx = fmaxf(mx, r[jj]);
      float se = 0.f;
#pragma unroll
      for (int jj = 0; jj < 16; ++jj) se += __expf(r[jj] - mx);
      bn = mx + __logf(se);                // new beta[i = lane's j]
      if (lane < 16) be[(size_t)t * 16 + j] = bn;
    }
  }
}

// ---- K6: per-(b,t) gamma/xi expected-LL contributions (one wave each) ----
__global__ __launch_bounds__(256)
void chmm_ll(const float* __restrict__ lt_all, const float* __restrict__ lev,
             const float* __restrict__ lsp, const float* __restrict__ alpha,
             const float* __restrict__ beta, const int* __restrict__ seq_len,
             float* __restrict__ contrib) {
  const int tid = threadIdx.x, wv = tid >> 5, lane = tid & 31, lq = lane & 15;
  const int bt = blockIdx.x * 8 + wv;
  const int b = bt / TT, t = bt % TT;
  const int len = seq_len[b];
  const int tr = (t + TT - len) % TT;  // beta rolled by (len - T)
  const float* evt = lev + (size_t)bt * 16;
  const float* btr = beta + ((size_t)b * TT + tr) * 16;
  const float at_ = alpha[(size_t)bt * 16 + lq];
  const float bt_ = btr[lq];
  float lg = at_ + bt_;
  float mx = rmax16(lg);
  float se = rsum16(__expf(lg - mx));
  float g = __expf(lg - (mx + __logf(se)));  // gamma
  float pr = 0.f, em = 0.f, tn = 0.f;
  if (t == 0) pr = rsum16(g * lsp[lq]) * (1.0f / BB);
  if (t < len) em = rsum16(g * evt[lq]) * (1.0f / BB);
  if (t >= 1 && t < len) {  // tmask: (t-1) < len-1
    const float* ltt = lt_all + (size_t)bt * 256;
    const float* ap = alpha + ((size_t)bt - 1) * 16;
    float x[8], l_[8];
#pragma unroll
    for (int q = 0; q < 8; ++q) {
      int idx = lane + 32 * q;
      int i = idx >> 4, j2 = idx & 15;
      l_[q] = ltt[idx];
      x[q] = l_[q] + ap[i] + evt[j2] + btr[j2];
    }
    float m2 = x[0];
#pragma unroll
    for (int q = 1; q < 8; ++q) m2 = fmaxf(m2, x[q]);
    m2 = rmax32(m2);
    float s2 = 0.f;
#pragma unroll
    for (int q = 0; q < 8; ++q) s2 += __expf(x[q] - m2);
    s2 = rsum32(s2);
    float norm = m2 + __logf(s2);
    float acc = 0.f;
#pragma unroll
    for (int q = 0; q < 8; ++q) acc += __expf(x[q] - norm) * l_[q];
    acc = rsum32(acc);
    tn = acc * (1.0f / BB);
  }
  if (lane == 0) contrib[bt] = pr + em + tn;
}

// ---- K7: deterministic reduction of contributions -> ll ----
__global__ __launch_bounds__(256)
void chmm_reduce(const float* __restrict__ contrib, float* __restrict__ out) {
  __shared__ float red[256];
  const int tid = threadIdx.x;
  float s = 0.f;
  for (int i = tid; i < BB * TT; i += 256) s += contrib[i];
  red[tid] = s;
  __syncthreads();
  for (int w = 128; w > 0; w >>= 1) {
    if (tid < w) red[tid] += red[tid + w];
    __syncthreads();
  }
  if (tid == 0) out[0] = red[0];
}

extern "C" void kernel_launch(void* const* d_in, const int* in_sizes, int n_in,
                              void* d_out, int out_size, void* d_ws, size_t ws_size,
                              hipStream_t stream) {
  (void)in_sizes; (void)n_in; (void)out_size; (void)ws_size;
  const float* emb = (const float*)d_in[0];
  const float* obs = (const float*)d_in[1];
  const int* seq_lengths = (const int*)d_in[2];
  const float* W_trans = (const float*)d_in[3];
  const float* b_trans = (const float*)d_in[4];
  const float* W_emiss = (const float*)d_in[5];
  const float* b_emiss = (const float*)d_in[6];
  const float* state_priors = (const float*)d_in[7];
  const float* unnorm_trans = (const float*)d_in[8];
  const float* unnorm_emiss = (const float*)d_in[9];
  float* out = (float*)d_out;
  char* ws = (char*)d_ws;

  _Float16* wt = (_Float16*)(ws + OFF_WT);
  float* bias = (float*)(ws + OFF_BIAS);
  float* tb = (float*)(ws + OFF_TB);
  float* eb = (float*)(ws + OFF_EB);
  float* lsp = (float*)(ws + OFF_LSP);
  float* lobs = (float*)(ws + OFF_LOBS);
  float* lev = (float*)(ws + OFF_LEV);
  float* alpha = (float*)(ws + OFF_ALPHA);
  float* beta = (float*)(ws + OFF_BETA);
  float* contrib = (float*)(ws + OFF_CONTRIB);

  float* out_trans = out + 1;                            // log_trans [B,T,16,16]
  float* out_emiss = out + 1 + (size_t)BB * TT * 256;    // log_emiss [B,T,10,16,16]

  const size_t nwt = (size_t)NMAT * 256 * DD;
  chmm_pack_w<<<(unsigned)((nwt + 255) / 256), 256, 0, stream>>>(W_trans, W_emiss, wt);
  chmm_prep_small<<<1, 256, 0, stream>>>(b_trans, b_emiss, unnorm_trans,
                                         unnorm_emiss, state_priors, bias, tb, eb, lsp);
  chmm_obs_prep<<<(BB * TT + 255) / 256, 256, 0, stream>>>(obs, lobs);
  chmm_gemm_fused<<<NROWT, 256, 0, stream>>>(emb, wt, bias, tb, eb, lobs,
                                             out_trans, out_emiss, lev);
  chmm_fb<<<2 * BB, 32, 0, stream>>>(out_trans, lev, lsp, alpha, beta);
  chmm_ll<<<(BB * TT) / 8, 256, 0, stream>>>(out_trans, lev, lsp, alpha, beta,
                                             seq_lengths, contrib);
  chmm_reduce<<<1, 256, 0, stream>>>(contrib, out);
}
